// ChameleonAttention_81578608820423
// MI455X (gfx1250) — compile-verified, matrix operand order checked by reading, not run on hardware
//
#include <hip/hip_runtime.h>
#include <hip/hip_bf16.h>

// ---------------- problem constants ----------------
#define T_SEQ   2048
#define HID     4096
#define NH      32
#define HD      128
#define QKV_N   12288          // (32 + 2*32) * 128
#define HALF    64
#define VT_P    40             // LDS pitch (bf16 elems) for transposed V tile; keeps b128 reads aligned

typedef __attribute__((ext_vector_type(16))) __bf16 bf16x16;
typedef __attribute__((ext_vector_type(8)))  __bf16 bf16x8;
typedef __attribute__((ext_vector_type(8)))  float  f32x8;

// ---------------------------------------------------------------------------
// WMMA helpers (CDNA5 wave32, V_WMMA_F32_16X16X32_BF16)
//
// A fragment (16x32, row-major source [row, K], ld in elements):
//   lane L: row = L%16 ; elems e=0..7  -> K = (L/16)*8 + e
//                        elems e=8..15 -> K = 16 + (L/16)*8 + (e-8)
// B fragment (32x16) from row-major [N, K] source uses the identical mapping
// with "row" = N.  Both halves are 8 contiguous bf16 -> two 16B loads.
// Works for global or LDS source pointers (ds_load_b128 when LDS).
// ---------------------------------------------------------------------------
__device__ __forceinline__ bf16x16 load_frag(const __bf16* __restrict__ p0, int ld, int lane) {
  const __bf16* p = p0 + (size_t)(lane & 15) * (size_t)ld + ((lane >> 4) << 3);
  bf16x8 lo = *(const bf16x8*)(p);
  bf16x8 hi = *(const bf16x8*)(p + 16);
  bf16x16 r;
#pragma unroll
  for (int i = 0; i < 8; ++i) { r[i] = lo[i]; r[i + 8] = hi[i]; }
  return r;
}

__device__ __forceinline__ f32x8 wmma_bf16(bf16x16 a, bf16x16 b, f32x8 c) {
  return __builtin_amdgcn_wmma_f32_16x16x32_bf16(false, a, false, b, (short)0, c, false, false);
}

__device__ __forceinline__ float redmax16(float x) {
#pragma unroll
  for (int m = 1; m <= 8; m <<= 1) x = fmaxf(x, __shfl_xor(x, m));
  return x;  // reduction stays within each 16-lane half (bit4 untouched)
}
__device__ __forceinline__ float redsum16(float x) {
#pragma unroll
  for (int m = 1; m <= 8; m <<= 1) x += __shfl_xor(x, m);
  return x;
}

// ---------------------------------------------------------------------------
// CDNA5 async global->LDS copy (ASYNCcnt tracked), per-lane b128.
// lds_off: 32-bit LDS byte offset (low 32 bits of generic shared address).
// ---------------------------------------------------------------------------
__device__ __forceinline__ void async_copy_b128(unsigned lds_off, const void* gaddr) {
  asm volatile("global_load_async_to_lds_b128 %0, %1, off"
               :: "v"(lds_off), "v"(gaddr) : "memory");
}
__device__ __forceinline__ void wait_async0() {
  asm volatile("s_wait_asynccnt 0x0" ::: "memory");
}

// ---------------------------------------------------------------------------
// f32 -> bf16 elementwise convert
// ---------------------------------------------------------------------------
__global__ __launch_bounds__(256) void f32_to_bf16_kernel(const float* __restrict__ in,
                                                          __bf16* __restrict__ out, int n) {
  int i = blockIdx.x * 256 + threadIdx.x;
  if (i < n) out[i] = (__bf16)in[i];
}

// ---------------------------------------------------------------------------
// C[M,N] (f32) = A[M,K] (bf16, row-major) * W[N,K]^T (bf16, row-major)
// 32x64 macro-tile per wave (2 A-frags x 4 B-frags -> 8 WMMAs / k-step),
// 8 waves in a 2(M) x 4(N) grid -> 64x256 block tile.
// ---------------------------------------------------------------------------
__global__ __launch_bounds__(256) void gemm_bf16_nt(const __bf16* __restrict__ A,
                                                    const __bf16* __restrict__ W,
                                                    float* __restrict__ C,
                                                    int M, int N, int K) {
  int lane = threadIdx.x & 31;
  int wv   = threadIdx.x >> 5;
  int wm   = wv >> 2;             // 0..1
  int wn   = wv & 3;              // 0..3
  int m0 = blockIdx.y * 64 + wm * 32;
  int n0 = blockIdx.x * 256 + wn * 64;

  const __bf16* ap0 = A + (size_t)m0 * (size_t)K;
  const __bf16* ap1 = A + (size_t)(m0 + 16) * (size_t)K;
  const __bf16* wp0 = W + (size_t)n0 * (size_t)K;
  const __bf16* wp1 = W + (size_t)(n0 + 16) * (size_t)K;
  const __bf16* wp2 = W + (size_t)(n0 + 32) * (size_t)K;
  const __bf16* wp3 = W + (size_t)(n0 + 48) * (size_t)K;

  f32x8 acc[2][4];
#pragma unroll
  for (int i = 0; i < 2; ++i)
#pragma unroll
    for (int j = 0; j < 4; ++j) acc[i][j] = (f32x8){};

  for (int k = 0; k < K; k += 32) {
    if (k + 1024 < K) {   // prefetch streaming panels ~32 iters ahead
      __builtin_prefetch(wp0 + k + 1024, 0, 3);
      __builtin_prefetch(ap0 + k + 1024, 0, 3);
    }
    bf16x16 a0 = load_frag(ap0 + k, K, lane);
    bf16x16 a1 = load_frag(ap1 + k, K, lane);
    bf16x16 b0 = load_frag(wp0 + k, K, lane);
    bf16x16 b1 = load_frag(wp1 + k, K, lane);
    bf16x16 b2 = load_frag(wp2 + k, K, lane);
    bf16x16 b3 = load_frag(wp3 + k, K, lane);
    acc[0][0] = wmma_bf16(a0, b0, acc[0][0]);
    acc[0][1] = wmma_bf16(a0, b1, acc[0][1]);
    acc[0][2] = wmma_bf16(a0, b2, acc[0][2]);
    acc[0][3] = wmma_bf16(a0, b3, acc[0][3]);
    acc[1][0] = wmma_bf16(a1, b0, acc[1][0]);
    acc[1][1] = wmma_bf16(a1, b1, acc[1][1]);
    acc[1][2] = wmma_bf16(a1, b2, acc[1][2]);
    acc[1][3] = wmma_bf16(a1, b3, acc[1][3]);
  }

  int g = lane >> 4, l = lane & 15;
#pragma unroll
  for (int i = 0; i < 2; ++i) {
#pragma unroll
    for (int j = 0; j < 4; ++j) {
      float* cp = C + (size_t)(m0 + i * 16 + g * 8) * (size_t)N + n0 + j * 16 + l;
#pragma unroll
      for (int v = 0; v < 8; ++v) cp[(size_t)v * (size_t)N] = acc[i][j][v];
    }
  }
}

// ---------------------------------------------------------------------------
// per-(t,head) LayerNorm(affine) + NeoX RoPE for Q and K.
// One wave per 128-elem row; lane owns 4 consecutive elements.
// Output layout: [head][t][128] bf16 (attention-friendly).
// ---------------------------------------------------------------------------
__global__ __launch_bounds__(256) void ln_rope_kernel(const float* __restrict__ qkv,
                                                      const int*   __restrict__ pos,
                                                      const float* __restrict__ qw,
                                                      const float* __restrict__ qb,
                                                      const float* __restrict__ kw,
                                                      const float* __restrict__ kb,
                                                      __bf16* __restrict__ qo,
                                                      __bf16* __restrict__ ko) {
  const float KF = 0.14391156856f;  // 2*ln(10000)/128
  int lane = threadIdx.x & 31;
  int r = blockIdx.x * 8 + (threadIdx.x >> 5);
  bool isQ = r < T_SEQ * NH;
  int rr = isQ ? r : r - T_SEQ * NH;
  int t = rr >> 5;
  int h = rr & 31;

  const float* src = qkv + (size_t)t * QKV_N + (isQ ? 0 : NH * HD) + h * HD;
  float4 x = *(const float4*)(src + lane * 4);

  float s  = x.x + x.y + x.z + x.w;
  float s2 = x.x * x.x + x.y * x.y + x.z * x.z + x.w * x.w;
#pragma unroll
  for (int m = 1; m <= 16; m <<= 1) { s += __shfl_xor(s, m); s2 += __shfl_xor(s2, m); }
  float mu   = s * (1.0f / HD);
  float var  = s2 * (1.0f / HD) - mu * mu;
  float rstd = rsqrtf(var + 1e-5f);

  const float* wrow = (isQ ? qw : kw) + h * HD;
  const float* brow = (isQ ? qb : kb) + h * HD;
  float4 wv = *(const float4*)(wrow + lane * 4);
  float4 bv = *(const float4*)(brow + lane * 4);

  float xn[4];
  xn[0] = (x.x - mu) * rstd * wv.x + bv.x;
  xn[1] = (x.y - mu) * rstd * wv.y + bv.y;
  xn[2] = (x.z - mu) * rstd * wv.z + bv.z;
  xn[3] = (x.w - mu) * rstd * wv.w + bv.w;

  float pt = (float)pos[t];
  __bf16* dst = (isQ ? qo : ko) + ((size_t)h * T_SEQ + t) * HD + lane * 4;
#pragma unroll
  for (int j = 0; j < 4; ++j) {
    int e = lane * 4 + j;
    int i = e & (HALF - 1);
    float ang = pt * __expf(-(float)i * KF);
    float sn, cs;
    __sincosf(ang, &sn, &cs);
    float other = __shfl_xor(xn[j], 16);  // partner between halves
    float o = (lane < 16) ? (xn[j] * cs - other * sn)   // x1*cos - x2*sin
                          : (xn[j] * cs + other * sn);  // x2*cos + x1*sin
    dst[j] = (__bf16)o;
  }
}

// ---------------------------------------------------------------------------
// V: slice + convert + re-layout to [head][t][128] bf16
// ---------------------------------------------------------------------------
__global__ __launch_bounds__(256) void v_convert_kernel(const float* __restrict__ qkv,
                                                        __bf16* __restrict__ vo) {
  int i = blockIdx.x * 256 + threadIdx.x;   // over T*4096
  int t = i >> 12;
  int c = i & 4095;
  int h = c >> 7;
  int d = c & 127;
  vo[((size_t)h * T_SEQ + t) * HD + d] = (__bf16)qkv[(size_t)t * QKV_N + 2 * NH * HD + c];
}

// ---------------------------------------------------------------------------
// Causal flash attention, one block per (128-query tile, head).
// 8 waves, each owning 16 query rows and a 16x128 f32 output accumulator.
// Key step = 32. Q fragments live in registers for the whole key loop;
// the shared 8KB K tile is staged into LDS with CDNA5 async-to-LDS copies.
// ---------------------------------------------------------------------------
__global__ __launch_bounds__(256) void flash_attn_kernel(const __bf16* __restrict__ Q,
                                                         const __bf16* __restrict__ Kb,
                                                         const __bf16* __restrict__ Vb,
                                                         __bf16* __restrict__ Ao) {
  __shared__ __align__(16) __bf16 Ks[32 * HD];        // [key][d] K tile (async-filled)
  __shared__ __align__(16) __bf16 Vt[HD * VT_P];      // [d][key] transposed V tile
  __shared__ __align__(16) __bf16 Pb[8 * 16 * 32];    // per-wave P tile, A-fragment layout

  const float scale = 0.08838834764831845f;  // 1/sqrt(128)
  int tid  = threadIdx.x;
  int lane = tid & 31;
  int wv   = tid >> 5;
  int g = lane >> 4, l = lane & 15;
  int h  = blockIdx.y;
  int qb = blockIdx.x;
  int qrow0 = qb * 128 + wv * 16;

  const __bf16* Qh = Q  + (size_t)h * T_SEQ * HD;
  const __bf16* Kh = Kb + (size_t)h * T_SEQ * HD;
  const __bf16* Vh = Vb + (size_t)h * T_SEQ * HD;

  // loop-invariant Q fragments (d = 0..127 in 4 k-steps)
  bf16x16 qf[4];
#pragma unroll
  for (int kk = 0; kk < 4; ++kk)
    qf[kk] = load_frag(Qh + (size_t)qrow0 * HD + kk * 32, HD, lane);

  float m_r[8], l_r[8], a_r[8];
  f32x8 O[8];
#pragma unroll
  for (int v = 0; v < 8; ++v) { m_r[v] = -1e30f; l_r[v] = 0.0f; }
#pragma unroll
  for (int c = 0; c < 8; ++c) O[c] = (f32x8){};

  unsigned ks_base = (unsigned)(uintptr_t)(void*)&Ks[0];

  int nkb = (qb + 1) * 4;  // key blocks of 32 covering [0, qb*128+128)
  for (int kblk = 0; kblk < nkb; ++kblk) {
    int key0 = kblk * 32;

    // ---- async-copy 32x128 K tile (8KB, contiguous) into LDS ----
    {
      const char* kg = (const char*)(Kh + (size_t)key0 * HD);
      async_copy_b128(ks_base + tid * 16,        kg + tid * 16);
      async_copy_b128(ks_base + tid * 16 + 4096, kg + tid * 16 + 4096);
    }

    // ---- cooperative load of 32x128 V tile, transposed into LDS ----
    {
      int key = tid >> 3;
      int d0  = (tid & 7) * 16;
      const __bf16* vp = Vh + (size_t)(key0 + key) * HD + d0;
      bf16x8 v0 = *(const bf16x8*)(vp);
      bf16x8 v1 = *(const bf16x8*)(vp + 8);
#pragma unroll
      for (int j = 0; j < 8; ++j) {
        Vt[(d0 + j) * VT_P + key]     = v0[j];
        Vt[(d0 + 8 + j) * VT_P + key] = v1[j];
      }
    }
    wait_async0();
    __syncthreads();

    // ---- S = Q K^T over d=128 (4 WMMA k-steps), two 16-key tiles ----
    f32x8 s0 = {}, s1 = {};
#pragma unroll
    for (int kk = 0; kk < 4; ++kk) {
      int k = kk * 32;
      bf16x16 b0 = load_frag(Ks + k,            HD, lane);
      bf16x16 b1 = load_frag(Ks + 16 * HD + k,  HD, lane);
      s0 = wmma_bf16(qf[kk], b0, s0);
      s1 = wmma_bf16(qf[kk], b1, s1);
    }

    // ---- online softmax update (row = g*8+v, col = l) ----
#pragma unroll
    for (int v = 0; v < 8; ++v) {
      int qg = qrow0 + g * 8 + v;
      float x0 = s0[v] * scale;
      float x1 = s1[v] * scale;
      if (key0 + l > qg)      x0 = -1e30f;
      if (key0 + 16 + l > qg) x1 = -1e30f;
      float rm = redmax16(fmaxf(x0, x1));
      float nm = fmaxf(m_r[v], rm);
      float p0 = __expf(x0 - nm);
      float p1 = __expf(x1 - nm);
      float rs = redsum16(p0 + p1);
      float alpha = __expf(m_r[v] - nm);
      l_r[v] = l_r[v] * alpha + rs;
      m_r[v] = nm;
      a_r[v] = alpha;
      __bf16* pp = Pb + wv * 512 + (g * 8 + v) * 32 + l;
      pp[0]  = (__bf16)p0;
      pp[16] = (__bf16)p1;
    }

    // ---- rescale running output, then O += P @ V ----
#pragma unroll
    for (int c = 0; c < 8; ++c) {
#pragma unroll
      for (int v = 0; v < 8; ++v) O[c][v] *= a_r[v];
    }
    bf16x16 pf = load_frag(Pb + wv * 512, 32, lane);
#pragma unroll
    for (int c = 0; c < 8; ++c) {
      bf16x16 vf = load_frag(Vt + (size_t)(c * 16) * VT_P, VT_P, lane);
      O[c] = wmma_bf16(pf, vf, O[c]);
    }
    __syncthreads();  // protect Ks/Vt before next iteration's overwrite
  }

  // ---- finalize and write attn [t][h*128 + d] as bf16 for output GEMM ----
#pragma unroll
  for (int v = 0; v < 8; ++v) {
    float inv = 1.0f / l_r[v];
    size_t row = (size_t)(qrow0 + g * 8 + v) * (NH * HD);
#pragma unroll
    for (int c = 0; c < 8; ++c) {
      Ao[row + h * HD + c * 16 + l] = (__bf16)(O[c][v] * inv);
    }
  }
}

// ---------------------------------------------------------------------------
// host-side launcher
// ---------------------------------------------------------------------------
extern "C" void kernel_launch(void* const* d_in, const int* in_sizes, int n_in,
                              void* d_out, int out_size, void* d_ws, size_t ws_size,
                              hipStream_t stream) {
  (void)in_sizes; (void)n_in; (void)out_size; (void)ws_size;
  const int*   positions = (const int*)d_in[0];
  const float* hs        = (const float*)d_in[1];
  const float* wqkv      = (const float*)d_in[2];
  const float* wo        = (const float*)d_in[3];
  const float* qnw       = (const float*)d_in[4];
  const float* qnb       = (const float*)d_in[5];
  const float* knw       = (const float*)d_in[6];
  const float* knb       = (const float*)d_in[7];
  float* out = (float*)d_out;

  char* ws = (char*)d_ws;
  size_t off = 0;
  __bf16* hsb   = (__bf16*)(ws + off); off += (size_t)T_SEQ * HID * 2;     // 16 MB
  __bf16* wqkvb = (__bf16*)(ws + off); off += (size_t)QKV_N * HID * 2;     // 96 MB
  __bf16* wob   = (__bf16*)(ws + off); off += (size_t)HID * HID * 2;       // 32 MB
  float*  qkvf  = (float*) (ws + off); off += (size_t)T_SEQ * QKV_N * 4;   // 96 MB
  __bf16* qbuf  = (__bf16*)(ws + off); off += (size_t)T_SEQ * HID * 2;     // 16 MB
  __bf16* kbuf  = (__bf16*)(ws + off); off += (size_t)T_SEQ * HID * 2;     // 16 MB
  __bf16* vbuf  = (__bf16*)(ws + off); off += (size_t)T_SEQ * HID * 2;     // 16 MB
  __bf16* attnb = (__bf16*)(ws + off); off += (size_t)T_SEQ * HID * 2;     // 16 MB

  // 1) downconvert activations and weights to bf16
  {
    int n = T_SEQ * HID;
    f32_to_bf16_kernel<<<(n + 255) / 256, 256, 0, stream>>>(hs, hsb, n);
  }
  {
    int n = QKV_N * HID;
    f32_to_bf16_kernel<<<(n + 255) / 256, 256, 0, stream>>>(wqkv, wqkvb, n);
  }
  {
    int n = HID * HID;
    f32_to_bf16_kernel<<<(n + 255) / 256, 256, 0, stream>>>(wo, wob, n);
  }

  // 2) QKV projection: [2048,4096] x [12288,4096]^T -> [2048,12288] f32
  gemm_bf16_nt<<<dim3(QKV_N / 256, T_SEQ / 64), 256, 0, stream>>>(
      hsb, wqkvb, qkvf, T_SEQ, QKV_N, HID);

  // 3) per-head LayerNorm + RoPE for Q and K -> bf16 [head][t][128]
  ln_rope_kernel<<<(T_SEQ * 2 * NH) / 8, 256, 0, stream>>>(
      qkvf, positions, qnw, qnb, knw, knb, qbuf, kbuf);

  // 4) V slice/convert -> bf16 [head][t][128]
  {
    int n = T_SEQ * HID;
    v_convert_kernel<<<n / 256, 256, 0, stream>>>(qkvf, vbuf);
  }

  // 5) causal flash attention -> attn bf16 [t][h*128+d]
  flash_attn_kernel<<<dim3(T_SEQ / 128, NH), 256, 0, stream>>>(qbuf, kbuf, vbuf, attnb);

  // 6) output projection: [2048,4096] x [4096,4096]^T -> f32 out
  gemm_bf16_nt<<<dim3(HID / 256, T_SEQ / 64), 256, 0, stream>>>(
      attnb, wob, out, T_SEQ, HID, HID);
}
